// RippleNet_66228395705122
// MI455X (gfx1250) — compile-verified, matrix operand order checked by reading, not run on hardware
//
#include <hip/hip_runtime.h>

typedef __attribute__((ext_vector_type(16))) _Float16 v16h;
typedef __attribute__((ext_vector_type(8)))  float    v8f;

#define B_TOT 2048
#define MM    32
#define DD    32
#define NHOP  2
#define NREL  32
#define WAVES 2
#define TB    16                       // batch rows per wave
#define ROWS_PER_BLOCK (WAVES * TB)    // 32
#define THREADS (WAVES * 32)           // 64

// K-index map for the 16-bit A/B WMMA fragment layout (wave32):
// lane<16 holds K in {0..7}U{16..23}, lane>=16 holds {8..15}U{24..31},
// VGPR p holds the f16 pair (K, K+1).
__device__ __forceinline__ int kmap(int p, int half) {
    return (p < 4) ? (half * 8 + 2 * p) : (16 + half * 8 + 2 * (p - 4));
}

__device__ __forceinline__ unsigned int pkh(float a, float b) {
    union { _Float16 h[2]; unsigned int u; } x;
    x.h[0] = (_Float16)a; x.h[1] = (_Float16)b;
    return x.u;
}

union Bf { uint4 q[2]; unsigned int u[8]; v16h h; };

__global__ __launch_bounds__(THREADS)
void ripplenet_kernel(const int* __restrict__ item_id,
                      const int* __restrict__ head_idx,
                      const int* __restrict__ rel_idx,
                      const int* __restrict__ tail_idx,
                      const float* __restrict__ ent,
                      const float* __restrict__ rel,
                      const float* __restrict__ Wt,
                      float* __restrict__ out)
{
    // Pre-packed B-matrix fragments (f16 pairs) for all 32 relation matrices
    __shared__ unsigned int sRfrag[NREL * 2][32][8];  // 64 KB, [kk][lane][p]
    __shared__ unsigned int sWfrag[2][32][8];         //  2 KB
    __shared__ _Float16     sQ[WAVES][NREL][TB][DD];  // 64 KB  Q[b,k,j]
    __shared__ float        sV[WAVES][TB][DD];        //  4 KB
    __shared__ float        sU[WAVES][TB][DD];        //  4 KB  user_emb
    __shared__ float        sO[WAVES][2][TB][DD];     //  8 KB  per-half o partials

    const int tid  = threadIdx.x;
    const int wave = tid >> 5;
    const int lane = tid & 31;
    const int r    = lane & 15;       // row within tile
    const int half = lane >> 4;       // which 16 columns / m-range this lane owns
    const int rowbase = blockIdx.x * ROWS_PER_BLOCK + wave * TB;
    const int brow = rowbase + r;

    // ---- pack relation matrices + transform_W into WMMA B-fragment layout ----
    for (int idx = tid; idx < NREL * 2 * 32 * 8; idx += THREADS) {
        int p = idx & 7, L = (idx >> 3) & 31, kk = idx >> 8;  // kk = k*2+n
        int j = ((kk & 1) << 4) | (L & 15);
        int i0 = kmap(p, L >> 4);
        const float* Rk = rel + (size_t)(kk >> 1) * DD * DD;
        sRfrag[kk][L][p] = pkh(Rk[i0 * DD + j], Rk[(i0 + 1) * DD + j]);
    }
    for (int idx = tid; idx < 2 * 32 * 8; idx += THREADS) {
        int p = idx & 7, L = (idx >> 3) & 31, n = (idx >> 8) & 1;
        int j = (n << 4) | (L & 15);
        int i0 = kmap(p, L >> 4);
        sWfrag[n][L][p] = pkh(Wt[i0 * DD + j], Wt[(i0 + 1) * DD + j]);
    }

    // ---- load v = entity_emb[item_id], init user accumulator ----
    {
        int iid = item_id[brow];
        const float4* vp = (const float4*)(ent + (size_t)iid * DD + half * 16);
        #pragma unroll
        for (int q = 0; q < 4; ++q) {
            float4 t = vp[q];
            int j = half * 16 + 4 * q;
            sV[wave][r][j + 0] = t.x; sV[wave][r][j + 1] = t.y;
            sV[wave][r][j + 2] = t.z; sV[wave][r][j + 3] = t.w;
            sU[wave][r][j + 0] = 0.f; sU[wave][r][j + 1] = 0.f;
            sU[wave][r][j + 2] = 0.f; sU[wave][r][j + 3] = 0.f;
        }
    }
    __syncthreads();

    const unsigned int* sRbase = &sRfrag[0][0][0];
    const v8f zc = {};

    for (int hop = 0; hop < NHOP; ++hop) {
        // ---- A fragment: current v tile (16x32), f32->f16 ----
        v16h av;
        #pragma unroll
        for (int p = 0; p < 8; ++p) {
            int i0 = kmap(p, half);
            av[2 * p + 0] = (_Float16)sV[wave][r][i0];
            av[2 * p + 1] = (_Float16)sV[wave][r][i0 + 1];
        }

        // ---- Q[b,k,:] = v_tile @ R_k for all 32 relations: 64 WMMAs ----
        // Software-pipelined: prefetch fragment kk+1 before WMMA kk; write
        // back result kk-1 after issuing WMMA kk (fills the WMMA->VALU
        // hazard slots with independent cvt/ds_store instead of v_nop).
        Bf cur, nxt;
        {
            const unsigned int* s0 = sRbase + (size_t)lane * 8;
            cur.q[0] = *(const uint4*)(s0);
            cur.q[1] = *(const uint4*)(s0 + 4);
            const unsigned int* s1 = sRbase + ((size_t)(32 + lane)) * 8;
            nxt.q[0] = *(const uint4*)(s1);
            nxt.q[1] = *(const uint4*)(s1 + 4);
        }
        v8f cp = __builtin_amdgcn_wmma_f32_16x16x32_f16(
                     false, av, false, cur.h, (short)0, zc, false, false);
        int kkp = 0;
        #pragma unroll 4
        for (int kk = 1; kk < 64; ++kk) {
            cur = nxt;
            const unsigned int* sn = sRbase + (size_t)((((kk + 1) & 63) * 32 + lane)) * 8;
            nxt.q[0] = *(const uint4*)(sn);
            nxt.q[1] = *(const uint4*)(sn + 4);
            v8f c = __builtin_amdgcn_wmma_f32_16x16x32_f16(
                        false, av, false, cur.h, (short)0, zc, false, false);
            // drain previous result (independent of the WMMA just issued)
            {
                int kp = kkp >> 1;
                int jp = ((kkp & 1) << 4) | (lane & 15);
                #pragma unroll
                for (int v = 0; v < 8; ++v)
                    sQ[wave][kp][v + 8 * half][jp] = (_Float16)cp[v];
            }
            cp = c; kkp = kk;
        }
        {
            int kp = kkp >> 1;
            int jp = ((kkp & 1) << 4) | (lane & 15);
            #pragma unroll
            for (int v = 0; v < 8; ++v)
                sQ[wave][kp][v + 8 * half][jp] = (_Float16)cp[v];
        }
        __syncthreads();

        // ---- probs[m] = Q[b, rel[m], :] . h[b,m]; 16 m's per lane ----
        float logit[16];
        float mx = -1e30f;
        const int ibase = (brow * NHOP + hop) * MM;
        #pragma unroll 4
        for (int mi = 0; mi < 16; ++mi) {
            int m = half * 16 + mi;
            int k = rel_idx[ibase + m];
            int hidx = head_idx[ibase + m];
            const float4* hp = (const float4*)(ent + (size_t)hidx * DD);
            const _Float16* qrow = &sQ[wave][k][r][0];
            float s = 0.f;
            #pragma unroll
            for (int q = 0; q < 8; ++q) {
                float4 hv = hp[q];
                s += (float)qrow[4 * q + 0] * hv.x + (float)qrow[4 * q + 1] * hv.y
                   + (float)qrow[4 * q + 2] * hv.z + (float)qrow[4 * q + 3] * hv.w;
            }
            logit[mi] = s;
            mx = fmaxf(mx, s);
        }
        // softmax across the two half-lanes owning this row
        mx = fmaxf(mx, __shfl_xor(mx, 16, 32));
        float e[16], ssum = 0.f;
        #pragma unroll
        for (int mi = 0; mi < 16; ++mi) { e[mi] = __expf(logit[mi] - mx); ssum += e[mi]; }
        ssum += __shfl_xor(ssum, 16, 32);
        float inv = 1.f / ssum;

        // ---- o[b,:] = sum_m probs[m] * t[b,m,:] ----
        float acc[32];
        #pragma unroll
        for (int j = 0; j < 32; ++j) acc[j] = 0.f;
        #pragma unroll 4
        for (int mi = 0; mi < 16; ++mi) {
            int m = half * 16 + mi;
            int tix = tail_idx[ibase + m];
            float p = e[mi] * inv;
            const float4* tp = (const float4*)(ent + (size_t)tix * DD);
            #pragma unroll
            for (int q = 0; q < 8; ++q) {
                float4 tv = tp[q];
                acc[4 * q + 0] += p * tv.x; acc[4 * q + 1] += p * tv.y;
                acc[4 * q + 2] += p * tv.z; acc[4 * q + 3] += p * tv.w;
            }
        }
        #pragma unroll
        for (int j = 0; j < 32; ++j) sO[wave][half][r][j] = acc[j];
        __syncthreads();

        // ---- v += o ; user += o ----
        #pragma unroll
        for (int jj = 0; jj < 16; ++jj) {
            int j = half * 16 + jj;
            float o = sO[wave][0][r][j] + sO[wave][1][r][j];
            sU[wave][r][j] += o;
            sV[wave][r][j] += o;
        }
        __syncthreads();

        // ---- v = (v + o) @ transform_W : 2 WMMAs ----
        v16h a2;
        #pragma unroll
        for (int p = 0; p < 8; ++p) {
            int i0 = kmap(p, half);
            a2[2 * p + 0] = (_Float16)sV[wave][r][i0];
            a2[2 * p + 1] = (_Float16)sV[wave][r][i0 + 1];
        }
        Bf w0, w1;
        w0.q[0] = *(const uint4*)&sWfrag[0][lane][0];
        w0.q[1] = *(const uint4*)&sWfrag[0][lane][4];
        w1.q[0] = *(const uint4*)&sWfrag[1][lane][0];
        w1.q[1] = *(const uint4*)&sWfrag[1][lane][4];
        v8f c0 = __builtin_amdgcn_wmma_f32_16x16x32_f16(
                     false, a2, false, w0.h, (short)0, zc, false, false);
        v8f c1 = __builtin_amdgcn_wmma_f32_16x16x32_f16(
                     false, a2, false, w1.h, (short)0, zc, false, false);
        {
            int j0 = (lane & 15);
            int j1 = 16 | (lane & 15);
            #pragma unroll
            for (int v = 0; v < 8; ++v) sV[wave][v + 8 * half][j0] = c0[v];
            #pragma unroll
            for (int v = 0; v < 8; ++v) sV[wave][v + 8 * half][j1] = c1[v];
        }
        __syncthreads();
    }

    // ---- score = sigmoid(v . user) ----
    float s = 0.f;
    #pragma unroll
    for (int jj = 0; jj < 16; ++jj) {
        int j = half * 16 + jj;
        s += sV[wave][r][j] * sU[wave][r][j];
    }
    s += __shfl_xor(s, 16, 32);
    if (half == 0)
        out[brow] = 1.f / (1.f + __expf(-s));
}

extern "C" void kernel_launch(void* const* d_in, const int* in_sizes, int n_in,
                              void* d_out, int out_size, void* d_ws, size_t ws_size,
                              hipStream_t stream) {
    const int*   item_id  = (const int*)d_in[0];
    const int*   head_idx = (const int*)d_in[1];
    const int*   rel_idx  = (const int*)d_in[2];
    const int*   tail_idx = (const int*)d_in[3];
    const float* ent      = (const float*)d_in[4];
    const float* rel      = (const float*)d_in[5];
    const float* Wt       = (const float*)d_in[6];
    float*       out      = (float*)d_out;

    dim3 grid(B_TOT / ROWS_PER_BLOCK);   // 64 blocks
    dim3 block(THREADS);                 // 64 threads = 2 waves
    ripplenet_kernel<<<grid, block, 0, stream>>>(item_id, head_idx, rel_idx,
                                                 tail_idx, ent, rel, Wt, out);
}
